// Mambaoptimizer_2817498546319
// MI455X (gfx1250) — matrix operand order repaired and checked by reading, not compiled, not run on hardware
//
#include <hip/hip_runtime.h>
#include <hip/hip_bf16.h>
#include <cstdint>

// ---------------------------------------------------------------------------
// Mamba stack forward for MI455X (gfx1250, wave32, WMMA).
// GEMMs: v_wmma_f32_16x16x32_f16, f16 operands staged via LDS, f32 accum.
// Two tile widths: wide (128x128 block, 8 WMMA/wave/chunk) for N%128==0,
// narrow (128x64) for the N=64 projections. Zero bounds checks in hot loops
// (host pads every GEMM operand).
// ---------------------------------------------------------------------------

typedef __attribute__((ext_vector_type(16))) _Float16 v16h;
typedef __attribute__((ext_vector_type(8)))  float    v8f;
typedef __attribute__((ext_vector_type(4)))  float    f4v;
typedef __attribute__((ext_vector_type(4)))  _Float16 h4v;

#define S_DIM   512
#define T_DIM   64
#define H_DIM   512
#define L_DIM   513                  // S + 1
#define B_DIM   64                   // = T
#define DI_DIM  1024
#define NS_DIM  16
#define R_DIM   32
#define FF_DIM  1408
#define NLAYER  4
#define TOKENS  (B_DIM * L_DIM)      // 32832
#define TOK_PAD 32896                // 257 * 128
#define EMB_TOK (S_DIM * T_DIM)      // 32768 = 256 * 128
#define KCONV   4
#define KEMB_P  64                   // embed K padded 48 -> 64

__device__ __forceinline__ float sigmoidf_(float x) { return 1.f / (1.f + __expf(-x)); }
__device__ __forceinline__ float siluf_(float x)    { return x * sigmoidf_(x); }
__device__ __forceinline__ float softplusf_(float x){ return (x > 20.f) ? x : log1pf(__expf(x)); }

// ---------------------------------------------------------------------------
// WMMA GEMM: C[M,N] = A[M,K] * W[N,K]^T (+bias[n]) (+act)
// Preconditions: M % 128 == 0, N % (32*NT) == 0, K % 32 == 0.
// 256 thr = 8 waves (4 in M x 2 in N). Each wave: 32 x (16*NT) output,
// 2 x NT WMMA accumulators, A fragments reused NT times.
// Software-pipelined: chunk k+1 global loads overlap chunk k WMMAs.
// act: 0 = none, 1 = softplus
// ---------------------------------------------------------------------------
#define TM  128
#define TKC 32

union AFragU { v16h v; _Float16 h[16]; };

template<int NT>   // N tiles (of 16) per wave: 2 -> TN=64, 4 -> TN=128
__global__ __launch_bounds__(256)
void gemm_wmma_f16(const float* __restrict__ A, int lda,
                   const float* __restrict__ W, int ldw,
                   float* __restrict__ C, int ldc,
                   int K, const float* __restrict__ bias, int act)
{
  constexpr int TN = 32 * NT;        // block tile N
  __shared__ _Float16 sA[TM][TKC + 8];
  __shared__ _Float16 sB[TN][TKC + 8];

  const int tid  = threadIdx.x;
  const int lane = tid & 31;
  const int wave = tid >> 5;
  const int wm   = wave >> 1;        // 0..3 (M)
  const int wn   = wave & 1;         // 0..1 (N)
  const int mBase = blockIdx.y * TM;
  const int nBase = blockIdx.x * TN;
  const int ml = lane & 15;
  const int hi = lane >> 4;

  // staging coords: A = 4 float4/thread (128x32), W = NT float4/thread (TNx32)
  const int ac = (tid & 7) * 4;
  int ar[4], wr[NT];
#pragma unroll
  for (int i = 0; i < 4; ++i)  ar[i] = (tid + i * 256) >> 3;
#pragma unroll
  for (int i = 0; i < NT; ++i) wr[i] = (tid + i * 256) >> 3;

  v8f acc[2][NT] = {};
  f4v ra[4], rw[NT];

  // prologue: chunk 0
#pragma unroll
  for (int i = 0; i < 4; ++i)
    ra[i] = *(const f4v*)(A + (size_t)(mBase + ar[i]) * lda + ac);
#pragma unroll
  for (int i = 0; i < NT; ++i)
    rw[i] = *(const f4v*)(W + (size_t)(nBase + wr[i]) * ldw + ac);

  const int nk = K / TKC;
  for (int kk = 0; kk < nk; ++kk) {
    // regs -> LDS (f16)
#pragma unroll
    for (int i = 0; i < 4; ++i)
      *(h4v*)(&sA[ar[i]][ac]) = __builtin_convertvector(ra[i], h4v);
#pragma unroll
    for (int i = 0; i < NT; ++i)
      *(h4v*)(&sB[wr[i]][ac]) = __builtin_convertvector(rw[i], h4v);
    __syncthreads();

    // issue next chunk's global loads (overlap with WMMA below)
    if (kk + 1 < nk) {
      const int k0 = (kk + 1) * TKC;
#pragma unroll
      for (int i = 0; i < 4; ++i)
        ra[i] = *(const f4v*)(A + (size_t)(mBase + ar[i]) * lda + k0 + ac);
#pragma unroll
      for (int i = 0; i < NT; ++i)
        rw[i] = *(const f4v*)(W + (size_t)(nBase + wr[i]) * ldw + k0 + ac);
      if (kk + 2 < nk)   // speculative prefetch of the chunk after next
        __builtin_prefetch(A + (size_t)(mBase + ar[0]) * lda + k0 + TKC + ac, 0, 1);
    }

    // fragments in contiguous k order (k = 16*half + j), same for A and B
    AFragU aF[2], bF[NT];
#pragma unroll
    for (int t = 0; t < 2; ++t) {
      // A 16x32: lane-half -> K-base 0/16; h[j] = Kbase + j
      const int rowA = wm * 32 + t * 16 + ml;
      const int kb   = hi * 16;
#pragma unroll
      for (int j = 0; j < 8; ++j)  aF[t].h[j] = sA[rowA][kb + j];
#pragma unroll
      for (int j = 8; j < 16; ++j) aF[t].h[j] = sA[rowA][kb + j];
    }
#pragma unroll
    for (int t = 0; t < NT; ++t) {
      // B 32x16: lane-half -> K-base 0/16; h[j] = Kbase + j
      const int rowB = wn * (16 * NT) + t * 16 + ml;
      const int kbB  = hi * 16;
#pragma unroll
      for (int j = 0; j < 16; ++j) bF[t].h[j] = sB[rowB][kbB + j];
    }
#pragma unroll
    for (int i = 0; i < 2; ++i)
#pragma unroll
      for (int j = 0; j < NT; ++j)
        acc[i][j] = __builtin_amdgcn_wmma_f32_16x16x32_f16(
            false, aF[i].v, false, bF[j].v, (short)0, acc[i][j], false, false);
    __syncthreads();
  }

  // epilogue: C/D layout -> lane n = lane&15, VGPR v -> m = v + (lane>>4)*8
#pragma unroll
  for (int i = 0; i < 2; ++i) {
#pragma unroll
    for (int j = 0; j < NT; ++j) {
      const int col = nBase + wn * (16 * NT) + j * 16 + ml;
      const float bv = bias ? bias[col] : 0.f;
#pragma unroll
      for (int v = 0; v < 8; ++v) {
        const int row = mBase + wm * 32 + i * 16 + hi * 8 + v;
        float x = acc[i][j][v] + bv;
        if (act == 1) x = softplusf_(x);
        C[(size_t)row * ldc + col] = x;
      }
    }
  }
}

// ---------------------------------------------------------------------------
// concat(x_x, x_y) -> [EMB_TOK, 64], zero-padded cols 48..63
// ---------------------------------------------------------------------------
__global__ void concat_kernel(const float* __restrict__ xx,
                              const float* __restrict__ xy,
                              float* __restrict__ cat, int n)
{
  int idx = blockIdx.x * blockDim.x + threadIdx.x;
  if (idx >= n) return;
  int tok = idx >> 6, f = idx & 63;
  float v = 0.f;
  if (f < 32)      v = xx[tok * 32 + f];
  else if (f < 48) v = xy[tok * 16 + (f - 32)];
  cat[idx] = v;
}

// x_emb_w [512,48] -> padded [512,64]
__global__ void pad_embw_kernel(const float* __restrict__ src,
                                float* __restrict__ dst, int n)
{
  int idx = blockIdx.x * blockDim.x + threadIdx.x;
  if (idx >= n) return;
  int r = idx >> 6, c = idx & 63;
  dst[idx] = (c < 48) ? src[r * 48 + c] : 0.f;
}

// ---------------------------------------------------------------------------
// hs[b][l][h]: l==0 -> x0[h], else emb[((l-1)*T + b)*H + h]
// ---------------------------------------------------------------------------
__global__ void build_hs_kernel(const float* __restrict__ emb,
                                const float* __restrict__ x0,
                                float* __restrict__ hs, int n)
{
  int idx = blockIdx.x * blockDim.x + threadIdx.x;
  if (idx >= n) return;
  int h = idx % H_DIM;
  int t = idx / H_DIM;
  int b = t / L_DIM, l = t % L_DIM;
  hs[idx] = (l == 0) ? x0[h] : emb[((size_t)(l - 1) * B_DIM + b) * H_DIM + h];
}

// ---------------------------------------------------------------------------
// residual = (add ? src + residual : src); out = LayerNorm(residual)*w + b
// ---------------------------------------------------------------------------
__global__ __launch_bounds__(256)
void resid_ln_kernel(const float* __restrict__ src, float* __restrict__ resid,
                     const float* __restrict__ w, const float* __restrict__ b,
                     float* __restrict__ out, int addFlag)
{
  __shared__ float s1[256], s2[256];
  const int tid = threadIdx.x;
  const size_t row = blockIdx.x;
  const float* s = src   + row * H_DIM;
  float*       r = resid + row * H_DIM;
  float v0 = s[tid], v1 = s[tid + 256];
  if (addFlag) { v0 += r[tid]; v1 += r[tid + 256]; }
  r[tid] = v0; r[tid + 256] = v1;
  s1[tid] = v0 + v1;
  s2[tid] = v0 * v0 + v1 * v1;
  __syncthreads();
  for (int off = 128; off > 0; off >>= 1) {
    if (tid < off) { s1[tid] += s1[tid + off]; s2[tid] += s2[tid + off]; }
    __syncthreads();
  }
  const float mu  = s1[0] * (1.f / H_DIM);
  const float var = s2[0] * (1.f / H_DIM) - mu * mu;
  const float inv = rsqrtf(var + 1e-5f);
  out[row * H_DIM + tid]       = (v0 - mu) * inv * w[tid]       + b[tid];
  out[row * H_DIM + tid + 256] = (v1 - mu) * inv * w[tid + 256] + b[tid + 256];
}

// ---------------------------------------------------------------------------
// causal depthwise conv (K=4) over l, + bias, + SiLU.
// ---------------------------------------------------------------------------
__global__ void conv_silu_kernel(const float* __restrict__ xz,
                                 const float* __restrict__ cw,
                                 const float* __restrict__ cb,
                                 float* __restrict__ xs, int n)
{
  int idx = blockIdx.x * blockDim.x + threadIdx.x;
  if (idx >= n) return;
  int d = idx % DI_DIM;
  int t = idx / DI_DIM;
  int b = t / L_DIM, l = t % L_DIM;
  float acc = cb[d];
#pragma unroll
  for (int k = 0; k < KCONV; ++k) {
    int ls = l - (KCONV - 1) + k;
    if (ls >= 0)
      acc += cw[d * KCONV + k] *
             xz[((size_t)b * L_DIM + ls) * (2 * DI_DIM) + d];
  }
  xs[(size_t)t * DI_DIM + d] = siluf_(acc);
}

// ---------------------------------------------------------------------------
// Selective scan. One block per batch element, 512 thr x 2 channels; state
// and A in registers; per-step B/C staged in LDS; fused (+x*D) * silu(z).
// ---------------------------------------------------------------------------
__global__ __launch_bounds__(512)
void scan_kernel(const float* __restrict__ dt,    // [*, DI]
                 const float* __restrict__ xs,    // [*, DI]
                 const float* __restrict__ dbc,   // [*, 64]
                 const float* __restrict__ xz,    // [*, 2*DI] (z at +DI)
                 const float* __restrict__ A_log, // [DI, NS]
                 const float* __restrict__ Dp,    // [DI]
                 float* __restrict__ ys)          // [*, DI]
{
  __shared__ float Bs[NS_DIM], Cs[NS_DIM];
  const int b   = blockIdx.x;
  const int tid = threadIdx.x;
  const int d0  = tid * 2;

  float Areg[2][NS_DIM], h[2][NS_DIM], Dreg[2];
#pragma unroll
  for (int c = 0; c < 2; ++c) {
    Dreg[c] = Dp[d0 + c];
#pragma unroll
    for (int n = 0; n < NS_DIM; ++n) {
      Areg[c][n] = -__expf(A_log[(size_t)(d0 + c) * NS_DIM + n]);
      h[c][n] = 0.f;
    }
  }

  for (int l = 0; l < L_DIM; ++l) {
    const size_t t = (size_t)b * L_DIM + l;
    if (tid < NS_DIM)          Bs[tid]          = dbc[t * 64 + R_DIM + tid];
    else if (tid < 2 * NS_DIM) Cs[tid - NS_DIM] = dbc[t * 64 + R_DIM + NS_DIM + (tid - NS_DIM)];
    __syncthreads();
#pragma unroll
    for (int c = 0; c < 2; ++c) {
      const int d = d0 + c;
      const float dtv = dt[t * DI_DIM + d];
      const float xv  = xs[t * DI_DIM + d];
      const float dx  = dtv * xv;
      float y = 0.f;
#pragma unroll
      for (int n = 0; n < NS_DIM; ++n) {
        const float dA = __expf(dtv * Areg[c][n]);
        h[c][n] = dA * h[c][n] + dx * Bs[n];
        y += h[c][n] * Cs[n];
      }
      y += xv * Dreg[c];
      const float zv = xz[t * (2 * DI_DIM) + DI_DIM + d];
      ys[t * DI_DIM + d] = y * siluf_(zv);
    }
    __syncthreads();
  }
}

// ---------------------------------------------------------------------------
// gated MLP combine: gated[t][f] = yg[t][f] * silu(yg[t][FF + f])
// ---------------------------------------------------------------------------
__global__ void gated_kernel(const float* __restrict__ yg,
                             float* __restrict__ gated, int n)
{
  int idx = blockIdx.x * blockDim.x + threadIdx.x;
  if (idx >= n) return;
  int f = idx % FF_DIM;
  int t = idx / FF_DIM;
  const float y = yg[(size_t)t * (2 * FF_DIM) + f];
  const float g = yg[(size_t)t * (2 * FF_DIM) + FF_DIM + f];
  gated[(size_t)t * FF_DIM + f] = y * siluf_(g);
}

// ---------------------------------------------------------------------------
// (B, L, Z) -> (L, B, Z)
// ---------------------------------------------------------------------------
__global__ void out_transpose_kernel(const float* __restrict__ z,
                                     float* __restrict__ out, int n)
{
  int idx = blockIdx.x * blockDim.x + threadIdx.x;
  if (idx >= n) return;
  int zc = idx & 63;
  int t  = idx >> 6;
  int b = t / L_DIM, l = t % L_DIM;
  out[((size_t)l * B_DIM + b) * 64 + zc] = z[idx];
}

// ---------------------------------------------------------------------------
// Host side
// ---------------------------------------------------------------------------
static void launch_gemm(const float* A, int lda, const float* W, int ldw,
                        float* C, int ldc, int M, int N, int K,
                        const float* bias, int act, hipStream_t stream)
{
  if (N % 128 == 0) {
    dim3 g((unsigned)(N / 128), (unsigned)(M / 128));
    gemm_wmma_f16<4><<<g, 256, 0, stream>>>(A, lda, W, ldw, C, ldc, K, bias, act);
  } else {
    dim3 g((unsigned)(N / 64), (unsigned)(M / 128));
    gemm_wmma_f16<2><<<g, 256, 0, stream>>>(A, lda, W, ldw, C, ldc, K, bias, act);
  }
}

extern "C" void kernel_launch(void* const* d_in, const int* in_sizes, int n_in,
                              void* d_out, int out_size, void* d_ws, size_t ws_size,
                              hipStream_t stream)
{
  (void)in_sizes; (void)n_in; (void)out_size; (void)ws_size;

  const float* x_x      = (const float*)d_in[0];
  const float* x_y      = (const float*)d_in[1];
  const float* x_emb_w  = (const float*)d_in[2];
  const float* x_emb_b  = (const float*)d_in[3];
  const float* x0       = (const float*)d_in[4];
  const float* norm1_w  = (const float*)d_in[5];
  const float* norm1_b  = (const float*)d_in[6];
  const float* in_proj  = (const float*)d_in[7];
  const float* conv_w   = (const float*)d_in[8];
  const float* conv_b   = (const float*)d_in[9];
  const float* x_proj   = (const float*)d_in[10];
  const float* dtp_w    = (const float*)d_in[11];
  const float* dtp_b    = (const float*)d_in[12];
  const float* A_log    = (const float*)d_in[13];
  const float* D_param  = (const float*)d_in[14];
  const float* out_proj = (const float*)d_in[15];
  const float* norm2_w  = (const float*)d_in[16];
  const float* norm2_b  = (const float*)d_in[17];
  const float* fc1_w    = (const float*)d_in[18];
  const float* fc2_w    = (const float*)d_in[19];
  const float* proj_w   = (const float*)d_in[20];
  const float* proj_b   = (const float*)d_in[21];
  float* out = (float*)d_out;

  // ---- workspace carve-up (floats); GEMM A/C buffers padded to TOK_PAD ----
  float* p = (float*)d_ws;
  float* cat   = p; p += (size_t)EMB_TOK * KEMB_P;
  float* embWp = p; p += (size_t)H_DIM * KEMB_P;
  float* emb   = p; p += (size_t)EMB_TOK * H_DIM;
  float* hs    = p; p += (size_t)TOK_PAD * H_DIM;
  float* resid = p; p += (size_t)TOK_PAD * H_DIM;
  float* ln    = p; p += (size_t)TOK_PAD * H_DIM;
  float* big   = p; p += (size_t)TOK_PAD * (2 * FF_DIM);  // xz (2048) / yg (2816)
  float* xs    = p; p += (size_t)TOK_PAD * FF_DIM;        // conv out (1024) / gated (1408)
  float* dbc   = p; p += (size_t)TOK_PAD * 64;
  float* dtb   = p; p += (size_t)TOK_PAD * DI_DIM;
  float* ysb   = p; p += (size_t)TOK_PAD * DI_DIM;
  float* mix   = p; p += (size_t)TOK_PAD * H_DIM;
  float* pout  = p; p += (size_t)TOK_PAD * 64;

  const int thr = 256;
  auto blks = [](long n, int t) { return (unsigned)((n + t - 1) / t); };

  // embedding
  {
    long n = (long)EMB_TOK * KEMB_P;
    concat_kernel<<<blks(n, thr), thr, 0, stream>>>(x_x, x_y, cat, (int)n);
    long nw = (long)H_DIM * KEMB_P;
    pad_embw_kernel<<<blks(nw, thr), thr, 0, stream>>>(x_emb_w, embWp, (int)nw);
    launch_gemm(cat, KEMB_P, embWp, KEMB_P, emb, H_DIM,
                EMB_TOK, H_DIM, KEMB_P, x_emb_b, 0, stream);
    long nh = (long)TOKENS * H_DIM;
    build_hs_kernel<<<blks(nh, thr), thr, 0, stream>>>(emb, x0, hs, (int)nh);
  }

  for (int i = 0; i < NLAYER; ++i) {
    const float* inw  = in_proj  + (size_t)i * (2 * DI_DIM) * H_DIM;
    const float* cwi  = conv_w   + (size_t)i * DI_DIM * KCONV;
    const float* cbi  = conv_b   + (size_t)i * DI_DIM;
    const float* xpw  = x_proj   + (size_t)i * 64 * DI_DIM;
    const float* dwi  = dtp_w    + (size_t)i * DI_DIM * R_DIM;
    const float* dbi  = dtp_b    + (size_t)i * DI_DIM;
    const float* ali  = A_log    + (size_t)i * DI_DIM * NS_DIM;
    const float* dpi  = D_param  + (size_t)i * DI_DIM;
    const float* opw  = out_proj + (size_t)i * H_DIM * DI_DIM;
    const float* f1w  = fc1_w    + (size_t)i * (2 * FF_DIM) * H_DIM;
    const float* f2w  = fc2_w    + (size_t)i * H_DIM * FF_DIM;

    // residual update + LN1
    resid_ln_kernel<<<TOKENS, 256, 0, stream>>>(
        hs, resid, norm1_w + i * H_DIM, norm1_b + i * H_DIM, ln, i > 0 ? 1 : 0);

    // in_proj: [TOK,512] x [2048,512]^T -> xz
    launch_gemm(ln, H_DIM, inw, H_DIM, big, 2 * DI_DIM,
                TOK_PAD, 2 * DI_DIM, H_DIM, nullptr, 0, stream);

    // depthwise causal conv + SiLU
    {
      long n = (long)TOKENS * DI_DIM;
      conv_silu_kernel<<<blks(n, thr), thr, 0, stream>>>(big, cwi, cbi, xs, (int)n);
    }

    // x_proj: [TOK,1024] x [64,1024]^T -> dbc   (narrow variant, N=64)
    launch_gemm(xs, DI_DIM, xpw, DI_DIM, dbc, 64,
                TOK_PAD, 64, DI_DIM, nullptr, 0, stream);

    // dt_proj + bias + softplus: dbc[:, :32] x [1024,32]^T
    launch_gemm(dbc, 64, dwi, R_DIM, dtb, DI_DIM,
                TOK_PAD, DI_DIM, R_DIM, dbi, 1, stream);

    // selective scan
    scan_kernel<<<B_DIM, 512, 0, stream>>>(dtb, xs, dbc, big, ali, dpi, ysb);

    // out_proj: [TOK,1024] x [512,1024]^T -> mix
    launch_gemm(ysb, DI_DIM, opw, DI_DIM, mix, H_DIM,
                TOK_PAD, H_DIM, DI_DIM, nullptr, 0, stream);

    // residual update + LN2
    resid_ln_kernel<<<TOKENS, 256, 0, stream>>>(
        mix, resid, norm2_w + i * H_DIM, norm2_b + i * H_DIM, ln, 1);

    // fc1: [TOK,512] x [2816,512]^T -> yg
    launch_gemm(ln, H_DIM, f1w, H_DIM, big, 2 * FF_DIM,
                TOK_PAD, 2 * FF_DIM, H_DIM, nullptr, 0, stream);

    // gated silu
    {
      long n = (long)TOKENS * FF_DIM;
      gated_kernel<<<blks(n, thr), thr, 0, stream>>>(big, xs, (int)n);
    }

    // fc2: [TOK,1408] x [512,1408]^T -> hs
    launch_gemm(xs, FF_DIM, f2w, FF_DIM, hs, H_DIM,
                TOK_PAD, H_DIM, FF_DIM, nullptr, 0, stream);
  }

  // final projection + bias (narrow variant, N=64)
  launch_gemm(hs, H_DIM, proj_w, H_DIM, pout, 64,
              TOK_PAD, 64, H_DIM, proj_b, 0, stream);

  // (B,L,Z) -> (L,B,Z)
  {
    long n = (long)TOKENS * 64;
    out_transpose_kernel<<<blks(n, thr), thr, 0, stream>>>(pout, out, (int)n);
  }
}
